// ConvNeXtGNN_31421980738085
// MI455X (gfx1250) — matrix operand order repaired and checked by reading, not compiled, not run on hardware
//
#include <hip/hip_runtime.h>
#include <hip/hip_bf16.h>

typedef unsigned short u16;
typedef unsigned int   u32;

// Problem constants (from reference)
#define B_    64
#define N_    256
#define D_    1024
#define HID_  512
#define NB_   2
#define NC_   1000
#define KNN_K 8
#define EPSLN 1e-5f

typedef __attribute__((ext_vector_type(16))) __bf16 v16bf;
typedef __attribute__((ext_vector_type(8)))  float  v8f;

union FragU { v16bf v; u32 u[8]; };

// -------- async global->LDS (gfx1250) with safe fallback --------------------
#if defined(__has_builtin)
#if __has_builtin(__builtin_amdgcn_global_load_async_to_lds_b128)
#define HAVE_ASYNC 1
#endif
#endif

typedef __attribute__((__vector_size__(16))) int v4i_;

__device__ __forceinline__ void cp16(const void* g, void* l) {
#ifdef HAVE_ASYNC
  __builtin_amdgcn_global_load_async_to_lds_b128(
      (__attribute__((address_space(1))) v4i_*)g,
      (__attribute__((address_space(3))) v4i_*)l, 0, 0);
#else
  *(uint4*)l = *(const uint4*)g;
#endif
}
// wait until ASYNCcnt <= n (async loads complete in order)
__device__ __forceinline__ void cp_wait4() {
#ifdef HAVE_ASYNC
  asm volatile("s_wait_asynccnt 0x4" ::: "memory");
#endif
}
__device__ __forceinline__ void cp_wait2() {
#ifdef HAVE_ASYNC
  asm volatile("s_wait_asynccnt 0x2" ::: "memory");
#endif
}
__device__ __forceinline__ void cp_wait0() {
#ifdef HAVE_ASYNC
  asm volatile("s_wait_asynccnt 0x0" ::: "memory");
#endif
}

// -------- scalar helpers ----------------------------------------------------
__device__ __forceinline__ u16 f2bf(float f) {
  u32 u = __float_as_uint(f);
  u32 r = u + 0x7FFFu + ((u >> 16) & 1u);   // RNE
  return (u16)(r >> 16);
}
__device__ __forceinline__ u32 pack2(float lo, float hi) {
  return (u32)f2bf(lo) | ((u32)f2bf(hi) << 16);
}
__device__ __forceinline__ float bf2f(u16 h) {
  return __uint_as_float(((u32)h) << 16);
}
__device__ __forceinline__ float gelu_f(float x) {
  return 0.5f * x * (1.0f + erff(x * 0.70710678118654752f));
}

// ---------------------------------------------------------------------------
// bf16 WMMA GEMM: C[M,N] = A[M,K] x B, f32 accum. 128x128x32 tile, 8 wave32,
// 2-stage LDS double buffering: async copies of tile t+1 overlap WMMA of t.
// A operand: bf16 row-major [M,K].
// BL=0: B given transposed, bf16 row-major [N,K]  (contiguous, async copy)
// BL=1: B given bf16 row-major [K,N]              (strided sync loader)
// EPI: 0 = +bias[col], write bf16 transposed  (lin1 -> H1T)
//      1 = gelu,       write bf16 transposed  (Amul1 -> H2T)
//      2 = +bias[row], write bf16 transposed  (lin2T -> T row-major)
//      3 = gram -> d2 f32
//      4 = residual: X += acc (f32)
// Transposed bf16 write: OT[((bout*N + col) * mt) + row], bout = bz + mrow/mt.
// ---------------------------------------------------------------------------
#define BM 128
#define BN 128
#define BK 32
#define STU 20   // dwords per LDS tile row (16 data + 4 pad; 80B keeps 16B align)

template<int BL, int EPI>
__global__ __launch_bounds__(256) void gemm_wmma_k(
    const u16* __restrict__ Ag, const u16* __restrict__ Bg,
    const float* __restrict__ bias, float* __restrict__ Cf,
    u16* __restrict__ Obf, float* __restrict__ Xg, const float* __restrict__ sqng,
    int M, int N, int K, int mt, long aB, long bB, long cB)
{
  __shared__ u32 la[2 * BM * STU];
  __shared__ u32 lb[2 * BN * STU];

  const int bz = blockIdx.z;
  const u16* A  = Ag + (long)bz * aB;
  const u16* Bm = Bg + (long)bz * bB;

  const int tid  = threadIdx.x;
  const int m0   = blockIdx.y * BM;
  const int n0   = blockIdx.x * BN;
  const int wave = tid >> 5, lane = tid & 31;
  const int wrow = wave >> 2, wcol = wave & 3;       // 2x4 wave grid
  const int lm   = lane & 15, half = lane >> 4;

  v8f acc[4][2];
#pragma unroll
  for (int mi = 0; mi < 4; mi++)
#pragma unroll
    for (int ni = 0; ni < 2; ni++)
#pragma unroll
      for (int r = 0; r < 8; r++) acc[mi][ni][r] = 0.0f;

  const int arow = tid >> 1;          // 0..127
  const int aseg = (tid & 1) * 16;    // k sub-segment (bf16 units)
  const int bnl  = tid & 127;         // strided-B: column
  const int bkh  = tid >> 7;          // strided-B: k half

  auto stage = [&](int kb, int buf) {
    {
      const u16* src = A + (long)(m0 + arow) * (long)K + kb + aseg;
      u32* dst = la + buf * (BM * STU) + arow * STU + (aseg >> 1);
      cp16(src, dst);
      cp16(src + 8, dst + 4);
    }
    if (BL == 0) {
      const u16* src = Bm + (long)(n0 + arow) * (long)K + kb + aseg;
      u32* dst = lb + buf * (BN * STU) + arow * STU + (aseg >> 1);
      cp16(src, dst);
      cp16(src + 8, dst + 4);
    } else {
      const u16* src = Bm + (long)(kb + bkh * 16) * (long)N + n0 + bnl;
      u32* dst = lb + buf * (BN * STU) + bnl * STU + bkh * 8;
#pragma unroll
      for (int q = 0; q < 8; q++) {
        u32 lo = src[(long)(2 * q) * N];
        u32 hi = src[(long)(2 * q + 1) * N];
        dst[q] = lo | (hi << 16);
      }
    }
  };

  const int nt = K / BK;
  stage(0, 0);                                   // prologue: tile 0 in flight

  for (int t = 0; t < nt; t++) {
    const int cur = t & 1;
    __syncthreads();                             // WAR: buffer cur^1 released
    if (t + 1 < nt) {
      stage((t + 1) * BK, cur ^ 1);              // issue next tile (overlaps)
      if (BL == 0) cp_wait4(); else cp_wait2();  // tile t's copies complete
    } else {
      cp_wait0();
    }
    __syncthreads();                             // tile t visible to all waves

    const u32* lap = la + cur * (BM * STU);
    const u32* lbp = lb + cur * (BN * STU);

    FragU af[4], bf[2];
#pragma unroll
    for (int mi = 0; mi < 4; mi++) {
      const u32* p = lap + (wrow * 64 + mi * 16 + lm) * STU;
#pragma unroll
      for (int r = 0; r < 8; r++) {
        int ko = (r & 3) + half * 4 + (r >> 2) * 8;  // A: K 0-7/16-23 | 8-15/24-31
        af[mi].u[r] = p[ko];
      }
    }
#pragma unroll
    for (int ni = 0; ni < 2; ni++) {
      const u32* p = lbp + (wcol * 32 + ni * 16 + lm) * STU;
#pragma unroll
      for (int r = 0; r < 8; r++) {
        int ko = (r & 3) + half * 4 + (r >> 2) * 8;  // B: K 0-7/16-23 | 8-15/24-31
        bf[ni].u[r] = p[ko];
      }
    }
#pragma unroll
    for (int mi = 0; mi < 4; mi++)
#pragma unroll
      for (int ni = 0; ni < 2; ni++)
        acc[mi][ni] = __builtin_amdgcn_wmma_f32_16x16x32_bf16(
            false, af[mi].v, false, bf[ni].v, (short)0, acc[mi][ni], false, false);
  }

  // ---- epilogues ------------------------------------------------------------
#pragma unroll
  for (int mi = 0; mi < 4; mi++) {
#pragma unroll
    for (int ni = 0; ni < 2; ni++) {
      const int nb = n0 + wcol * 32 + ni * 16 + lm;                 // column
      const int mb = m0 + wrow * 64 + mi * 16 + half * 8;           // row base (8 consec)
      if (EPI <= 2) {
        const int bout = bz + mb / mt;
        const int ml   = mb % mt;
        u32 pk[4];
#pragma unroll
        for (int r2 = 0; r2 < 4; r2++) {
          float v0 = acc[mi][ni][2 * r2 + 0];
          float v1 = acc[mi][ni][2 * r2 + 1];
          if (EPI == 0) { v0 += bias[nb]; v1 += bias[nb]; }
          else if (EPI == 1) { v0 = gelu_f(v0); v1 = gelu_f(v1); }
          else { v0 += bias[ml + 2 * r2]; v1 += bias[ml + 2 * r2 + 1]; }
          pk[r2] = pack2(v0, v1);
        }
        long oidx = ((long)bout * N + nb) * (long)mt + ml;          // 16B aligned
        *(uint4*)(Obf + oidx) = *(const uint4*)pk;
      } else if (EPI == 3) {
        const float* sqn = sqng + (long)bz * N_;
        float* C = Cf + (long)bz * cB;
#pragma unroll
        for (int r = 0; r < 8; r++) {
          int mrow = mb + r;
          float d2 = sqn[mrow] + sqn[nb] - 2.0f * acc[mi][ni][r];
          if (mrow == nb) d2 = 3.0e38f;
          C[(long)mrow * N + nb] = d2;
        }
      } else {  // EPI == 4: residual into f32 X
        float* X = Xg + (long)bz * cB;
#pragma unroll
        for (int r = 0; r < 8; r++) {
          int mrow = mb + r;
          X[(long)mrow * N + nb] += acc[mi][ni][r];
        }
      }
    }
  }
}

// ---------------------------------------------------------------------------
// LayerNorm over D=1024. MODE 0: Outh = bf16(LN(X)*g+b)
//                        MODE 1: Outf = X + gelu(LN(X)*g+b)   (f32, in place)
//                        MODE 2: Outh = bf16(gelu(LN(X)*g+b))
// ---------------------------------------------------------------------------
template<int MODE>
__global__ __launch_bounds__(256) void ln_k(
    const float* __restrict__ Xin, float* __restrict__ Outf, u16* __restrict__ Outh,
    const float* __restrict__ g, const float* __restrict__ be)
{
  __shared__ float red[256];
  const long row = blockIdx.x;
  const float* x = Xin + row * (long)D_;
  const int tid = threadIdx.x;

  float vals[4]; float s = 0.0f;
#pragma unroll
  for (int i = 0; i < 4; i++) { vals[i] = x[tid + i * 256]; s += vals[i]; }
  red[tid] = s; __syncthreads();
  for (int o = 128; o > 0; o >>= 1) { if (tid < o) red[tid] += red[tid + o]; __syncthreads(); }
  const float mean = red[0] * (1.0f / D_);
  __syncthreads();
  float s2 = 0.0f;
#pragma unroll
  for (int i = 0; i < 4; i++) { float d = vals[i] - mean; s2 += d * d; }
  red[tid] = s2; __syncthreads();
  for (int o = 128; o > 0; o >>= 1) { if (tid < o) red[tid] += red[tid + o]; __syncthreads(); }
  const float inv = rsqrtf(red[0] * (1.0f / D_) + EPSLN);

#pragma unroll
  for (int i = 0; i < 4; i++) {
    const int c = tid + i * 256;
    float h = (vals[i] - mean) * inv * g[c] + be[c];
    if (MODE == 0)      Outh[row * (long)D_ + c] = f2bf(h);
    else if (MODE == 1) Outf[row * (long)D_ + c] = vals[i] + gelu_f(h);
    else                Outh[row * (long)D_ + c] = f2bf(gelu_f(h));
  }
}

// Row squared-norm over D=1024 (block per row)
__global__ __launch_bounds__(256) void rownorm_k(const float* __restrict__ X,
                                                 float* __restrict__ sqn)
{
  __shared__ float red[256];
  const long row = blockIdx.x;
  const float* x = X + row * (long)D_;
  const int tid = threadIdx.x;
  float s = 0.0f;
#pragma unroll
  for (int i = 0; i < 4; i++) { float v = x[tid + i * 256]; s += v * v; }
  red[tid] = s; __syncthreads();
  for (int o = 128; o > 0; o >>= 1) { if (tid < o) red[tid] += red[tid + o]; __syncthreads(); }
  if (tid == 0) sqn[row] = red[0];
}

// flat f32 -> bf16 convert (4 elems/thread)
__global__ __launch_bounds__(256) void cvt_k(const float* __restrict__ S,
                                             u16* __restrict__ Dst, long n4)
{
  long i = (long)blockIdx.x * blockDim.x + threadIdx.x;
  if (i >= n4) return;
  float4 f = ((const float4*)S)[i];
  uint2 o; o.x = pack2(f.x, f.y); o.y = pack2(f.z, f.w);
  ((uint2*)Dst)[i] = o;
}

// transpose + convert: S[R,C] f32 -> T[C,R] bf16 (LDS 32x32 tile)
__global__ __launch_bounds__(256) void tc_k(const float* __restrict__ S,
                                            u16* __restrict__ T, int R, int C)
{
  __shared__ float t[32][33];
  const int c0 = blockIdx.x * 32, r0 = blockIdx.y * 32;
  const int tx = threadIdx.x & 31, ty = threadIdx.x >> 5;
#pragma unroll
  for (int p = 0; p < 4; p++)
    t[ty + 8 * p][tx] = S[(long)(r0 + ty + 8 * p) * C + c0 + tx];
  __syncthreads();
#pragma unroll
  for (int p = 0; p < 4; p++)
    T[(long)(c0 + ty + 8 * p) * R + r0 + tx] = f2bf(t[tx][ty + 8 * p]);
}

// Per-row 8-NN selection -> symmetric adjacency (+ self loop). Block per row.
__global__ __launch_bounds__(256) void topk_k(const float* __restrict__ d2,
                                              float* __restrict__ Aadj)
{
  __shared__ float rowv[256];
  __shared__ float v[256];
  __shared__ int   vi[256];
  const long row = blockIdx.x;
  const int b = (int)(row / N_), i = (int)(row % N_);
  const int tid = threadIdx.x;
  rowv[tid] = (tid == i) ? 3.4e38f : d2[row * (long)N_ + tid];
  __syncthreads();
  float* Ab = Aadj + (long)b * N_ * N_;
  for (int t = 0; t < KNN_K; t++) {
    v[tid] = rowv[tid]; vi[tid] = tid; __syncthreads();
    for (int o = 128; o > 0; o >>= 1) {
      if (tid < o) {
        if (v[tid + o] < v[tid] || (v[tid + o] == v[tid] && vi[tid + o] < vi[tid])) {
          v[tid] = v[tid + o]; vi[tid] = vi[tid + o];
        }
      }
      __syncthreads();
    }
    if (tid == 0) {
      int j = vi[0];
      rowv[j] = 3.4e38f;
      Ab[(long)i * N_ + j] = 1.0f;
      Ab[(long)j * N_ + i] = 1.0f;
    }
    __syncthreads();
  }
  if (tid == 0) Ab[(long)i * N_ + i] = 1.0f;
}

// dinv = rsqrt(rowsum(A)) on unscaled adjacency
__global__ __launch_bounds__(256) void deg_k(const float* __restrict__ Aadj,
                                             float* __restrict__ dinv)
{
  const long row = (long)blockIdx.x * blockDim.x + threadIdx.x;
  if (row >= (long)B_ * N_) return;
  const float* r = Aadj + row * (long)N_;
  float s = 0.0f;
  for (int j = 0; j < N_; j++) s += r[j];
  dinv[row] = rsqrtf(s);
}

// A_bf = bf16(A * dinv_i * dinv_j)  (symmetric -> serves as both A and A^T)
__global__ __launch_bounds__(256) void scale_k(const float* __restrict__ Aadj,
                                               const float* __restrict__ dinv,
                                               u16* __restrict__ Abf)
{
  const long idx = (long)blockIdx.x * blockDim.x + threadIdx.x;
  if (idx >= (long)B_ * N_ * N_) return;
  const long nn = (long)N_ * N_;
  const int b = (int)(idx / nn);
  const int rem = (int)(idx % nn);
  const int i = rem / N_, j = rem % N_;
  Abf[idx] = f2bf(Aadj[idx] * dinv[(long)b * N_ + i] * dinv[(long)b * N_ + j]);
}

// mean pool over nodes from bf16 activations
__global__ __launch_bounds__(256) void pool_k(const u16* __restrict__ T,
                                              float* __restrict__ pool)
{
  const long idx = (long)blockIdx.x * blockDim.x + threadIdx.x;
  if (idx >= (long)B_ * D_) return;
  const int b = (int)(idx / D_), d = (int)(idx % D_);
  const u16* p = T + (long)b * N_ * D_ + d;
  float s = 0.0f;
  for (int n = 0; n < N_; n++) s += bf2f(p[(long)n * D_]);
  pool[idx] = s * (1.0f / N_);
}

// logits = pool @ Wc + bc (tiny; f32 VALU)
__global__ __launch_bounds__(256) void cls_k(const float* __restrict__ pool,
                                             const float* __restrict__ Wc,
                                             const float* __restrict__ bc,
                                             float* __restrict__ out)
{
  const long idx = (long)blockIdx.x * blockDim.x + threadIdx.x;
  if (idx >= (long)B_ * NC_) return;
  const int b = (int)(idx / NC_), c = (int)(idx % NC_);
  const float* p = pool + (long)b * D_;
  float s = bc[c];
  for (int d = 0; d < D_; d++) s += p[d] * Wc[(long)d * NC_ + c];
  out[idx] = s;
}

// ---------------------------------------------------------------------------
extern "C" void kernel_launch(void* const* d_in, const int* in_sizes, int n_in,
                              void* d_out, int out_size, void* d_ws, size_t ws_size,
                              hipStream_t stream)
{
  (void)in_sizes; (void)n_in; (void)out_size; (void)ws_size;
  const float* tokens = (const float*)d_in[0];
  const float* W1  = (const float*)d_in[1];
  const float* b1  = (const float*)d_in[2];
  const float* W2  = (const float*)d_in[3];
  const float* b2  = (const float*)d_in[4];
  const float* g1  = (const float*)d_in[5];
  const float* be1 = (const float*)d_in[6];
  const float* g2  = (const float*)d_in[7];
  const float* be2 = (const float*)d_in[8];
  const float* gr  = (const float*)d_in[9];
  const float* br  = (const float*)d_in[10];
  const float* Wc  = (const float*)d_in[11];
  const float* bc  = (const float*)d_in[12];
  float* out = (float*)d_out;

  char* ws = (char*)d_ws;
  size_t off = 0;
  float* X     = (float*)(ws + off); off += (size_t)B_ * N_ * D_ * 4;   // 64 MB
  float* Aadj  = (float*)(ws + off); off += (size_t)B_ * N_ * N_ * 4;   // 16 MB
  float* d2    = (float*)(ws + off); off += (size_t)B_ * N_ * N_ * 4;   // 16 MB
  u16*   Abf   = (u16*)  (ws + off); off += (size_t)B_ * N_ * N_ * 2;   //  8 MB
  u16*   tokbf = (u16*)  (ws + off); off += (size_t)B_ * N_ * D_ * 2;   // 32 MB
  u16*   Tbf   = (u16*)  (ws + off); off += (size_t)B_ * N_ * D_ * 2;   // 32 MB
  u16*   H1T   = (u16*)  (ws + off); off += (size_t)B_ * HID_ * N_ * 2; // 16 MB
  u16*   H2T   = (u16*)  (ws + off); off += (size_t)B_ * HID_ * N_ * 2; // 16 MB
  u16*   W1T   = (u16*)  (ws + off); off += (size_t)D_ * HID_ * 2;      //  1 MB
  u16*   W2T   = (u16*)  (ws + off); off += (size_t)D_ * HID_ * 2;      //  1 MB
  float* sqn   = (float*)(ws + off); off += (size_t)B_ * N_ * 4;
  float* pool  = (float*)(ws + off); off += (size_t)B_ * D_ * 4;

  // x = tokens (mutated by residuals); tokens -> bf16 once
  (void)hipMemcpyAsync(X, tokens, (size_t)B_ * N_ * D_ * 4, hipMemcpyDeviceToDevice, stream);
  cvt_k<<<(int)((long)B_ * N_ * D_ / 4 / 256), 256, 0, stream>>>(tokens, tokbf,
                                                                 (long)B_ * N_ * D_ / 4);
  // kNN adjacency
  rownorm_k<<<B_ * N_, 256, 0, stream>>>(tokens, sqn);
  gemm_wmma_k<0, 3><<<dim3(N_ / BN, N_ / BM, B_), 256, 0, stream>>>(
      tokbf, tokbf, nullptr, d2, nullptr, nullptr, sqn,
      N_, N_, D_, 256, (long)N_ * D_, (long)N_ * D_, (long)N_ * N_);
  (void)hipMemsetAsync(Aadj, 0, (size_t)B_ * N_ * N_ * 4, stream);
  topk_k<<<B_ * N_, 256, 0, stream>>>(d2, Aadj);
  deg_k<<<(B_ * N_) / 256, 256, 0, stream>>>(Aadj, sqn);   // sqn reused as dinv
  scale_k<<<(B_ * N_ * N_) / 256, 256, 0, stream>>>(Aadj, sqn, Abf);

  for (int i = 0; i < NB_; i++) {
    // weights -> transposed bf16 (per block)
    tc_k<<<dim3(HID_ / 32, D_ / 32), 256, 0, stream>>>(W1 + (size_t)i * D_ * HID_, W1T, D_, HID_);
    tc_k<<<dim3(D_ / 32, HID_ / 32), 256, 0, stream>>>(W2 + (size_t)i * HID_ * D_, W2T, HID_, D_);
    // Tbf = bf16(LN1(x))
    ln_k<0><<<B_ * N_, 256, 0, stream>>>(X, nullptr, Tbf, g1 + (size_t)i * D_, be1 + (size_t)i * D_);
    // H1T[b,h,n] = (Tbf @ W1 + b1)^T
    gemm_wmma_k<0, 0><<<dim3(HID_ / BN, (B_ * N_) / BM, 1), 256, 0, stream>>>(
        Tbf, W1T, b1 + (size_t)i * HID_, nullptr, H1T, nullptr, nullptr,
        B_ * N_, HID_, D_, 256, 0, 0, 0);
    // H2T[b,h,n] = gelu(A @ H1)^T
    gemm_wmma_k<0, 1><<<dim3(HID_ / BN, N_ / BM, B_), 256, 0, stream>>>(
        Abf, H1T, nullptr, nullptr, H2T, nullptr, nullptr,
        N_, HID_, N_, 256, (long)N_ * N_, (long)HID_ * N_, 0);
    // T[b,n,d] = (W2^T @ H2^T)^T + b2 = H2 @ W2 + b2   (computed transposed)
    gemm_wmma_k<1, 2><<<dim3(N_ / BN, D_ / BM, B_), 256, 0, stream>>>(
        W2T, H2T, b2 + (size_t)i * D_, nullptr, Tbf, nullptr, nullptr,
        D_, N_, HID_, 1024, 0, (long)HID_ * N_, 0);
    // x += A @ T
    gemm_wmma_k<1, 4><<<dim3(D_ / BN, N_ / BM, B_), 256, 0, stream>>>(
        Abf, Tbf, nullptr, nullptr, nullptr, X, nullptr,
        N_, D_, N_, 256, (long)N_ * N_, (long)N_ * D_, (long)N_ * D_);
    // x += gelu(LN2(x))
    ln_k<1><<<B_ * N_, 256, 0, stream>>>(X, X, nullptr, g2 + (size_t)i * D_, be2 + (size_t)i * D_);
  }

  // readout: Tbf = bf16(gelu(LN(x))); mean pool; classifier
  ln_k<2><<<B_ * N_, 256, 0, stream>>>(X, nullptr, Tbf, gr, br);
  pool_k<<<(B_ * D_) / 256, 256, 0, stream>>>(Tbf, pool);
  cls_k<<<(B_ * NC_ + 255) / 256, 256, 0, stream>>>(pool, Wc, bc, out);
}
